// Attention_73203422593618
// MI455X (gfx1250) — matrix operand order repaired and checked by reading, not compiled, not run on hardware
//
#include <hip/hip_runtime.h>
#include <hip/hip_bf16.h>

// ---------------- CDNA5 WMMA types ----------------
typedef __attribute__((ext_vector_type(16))) __bf16 v16bf;
typedef __attribute__((ext_vector_type(2)))  __bf16 v2bf;
typedef __attribute__((ext_vector_type(8)))  float  v8f;
typedef __attribute__((ext_vector_type(2)))  float  v2f;

union Frag { unsigned int u[8]; v16bf v; };

// packed f32x2 -> bf16x2 (targets v_cvt_pk_bf16_f32 on gfx1250)
__device__ __forceinline__ unsigned pack2bf(float a, float b) {
    v2f f = {a, b};
    union { v2bf h; unsigned u; } u;
    u.h = __builtin_convertvector(f, v2bf);
    return u.u;
}
__device__ __forceinline__ unsigned short f2bf(float f) {
    union { __bf16 h; unsigned short s; } u;
    u.h = (__bf16)f;
    return u.s;
}

__device__ __forceinline__ v8f wmma_bf16(v16bf a, v16bf b, v8f c) {
    return __builtin_amdgcn_wmma_f32_16x16x32_bf16(false, a, false, b,
                                                   (short)0, c, false, false);
}

// ---- gfx1250 async global->LDS copy (ASYNCcnt path) ----
__device__ __forceinline__ unsigned lds_addr32(const void* p) {
    return (unsigned)(unsigned long long)
        (__attribute__((address_space(3))) const void*)p;
}
__device__ __forceinline__ void async_ld_b128(unsigned lds_off, const void* gptr) {
    asm volatile("global_load_async_to_lds_b128 %0, %1, off"
                 :: "v"(lds_off), "v"(gptr) : "memory");
}
__device__ __forceinline__ void wait_asynccnt0() {
    asm volatile("s_wait_asynccnt 0x0" ::: "memory");
}

// ---------------- problem sizes ----------------
#define BB 4
#define S0 768
#define S1 256
#define TT 1024
#define W0 2048
#define W1 1024
#define NH 8
#define HH 256
#define BIGNEG (-3.0e38f)

// ============================================================
// Generic tiled GEMM:  C[M,Nc] = A[M,Kd] (f32) @ B[Kd,Nc] (f32)
// bf16 WMMA 16x16x32, f32 accumulate. 64x64x32 tiles, 256 thr.
// ============================================================
__global__ __launch_bounds__(256) void gemm_bf16_kernel(
    const float* __restrict__ A, const float* __restrict__ Bm, float* __restrict__ C,
    int M, int Nc, int Kd, int lda, int ldb, int ldc,
    long sAb, long sBn, long sCb, long sCn, int nInner)
{
    int z  = blockIdx.z;
    int bb = z / nInner;
    int nn = z - bb * nInner;
    A  += (long)bb * sAb;
    Bm += (long)nn * sBn;
    C  += (long)bb * sCb + (long)nn * sCn;

    __shared__ unsigned short As[64][36];   // [row][k], padded (72B row)
    __shared__ unsigned short Bt[64][36];   // [col][k] (B transposed)

    const int tid  = threadIdx.x;
    const int lane = tid & 31;
    const int w    = tid >> 5;
    const int wm   = w >> 2;
    const int wn   = w & 3;
    const int l16  = lane & 15;
    const int hi   = lane >> 4;

    const int row0 = blockIdx.y * 64;
    const int col0 = blockIdx.x * 64;

    v8f c0 = (v8f){}; v8f c1 = (v8f){};

    const int ar = tid >> 2;
    const int ac = (tid & 3) * 8;
    const int bc = tid & 63;
    const int bk = (tid >> 6) * 8;

    const float* ga = A + (long)(row0 + ar) * lda + ac;
    const float* gb0 = Bm + (long)bk * ldb + col0 + bc;

    for (int kk = 0; kk < Kd; kk += 32) {
        #pragma unroll
        for (int j = 0; j < 4; ++j)
            *(unsigned*)&As[ar][ac + 2 * j] = pack2bf(ga[2 * j], ga[2 * j + 1]);
        ga += 32;

        const float* gb = gb0;
        #pragma unroll
        for (int j = 0; j < 4; ++j) {
            *(unsigned*)&Bt[bc][bk + 2 * j] = pack2bf(gb[0], gb[(long)ldb]);
            gb += 2L * ldb;
        }
        gb0 += 32L * ldb;
        __syncthreads();

        Frag a0, a1, bfr;
        const unsigned short* ap0 = &As[wm * 32 + l16][0];
        const unsigned short* ap1 = &As[wm * 32 + 16 + l16][0];
        const unsigned short* bp  = &Bt[wn * 16 + l16][0];
        const int lo  = hi * 8;
        const int klo = hi * 16;
        #pragma unroll
        for (int j = 0; j < 4; ++j) {
            a0.u[j]     = *(const unsigned int*)(ap0 + lo + 2 * j);
            a0.u[4 + j] = *(const unsigned int*)(ap0 + 16 + lo + 2 * j);
            a1.u[j]     = *(const unsigned int*)(ap1 + lo + 2 * j);
            a1.u[4 + j] = *(const unsigned int*)(ap1 + 16 + lo + 2 * j);
        }
        #pragma unroll
        for (int j = 0; j < 8; ++j)
            bfr.u[j] = *(const unsigned int*)(bp + lo + 2 * j + (j >= 4 ? 8 : 0));

        c0 = wmma_bf16(a0.v, bfr.v, c0);
        c1 = wmma_bf16(a1.v, bfr.v, c1);
        __syncthreads();
    }

    // epilogue: incremental row pointers (no per-store 64-bit mul)
    float* cp0 = C + (long)(row0 + wm * 32 + hi * 8) * ldc + col0 + wn * 16 + l16;
    float* cp1 = cp0 + 16L * ldc;
    #pragma unroll
    for (int j = 0; j < 8; ++j) {
        *cp0 = c0[j]; *cp1 = c1[j];
        cp0 += ldc; cp1 += ldc;
    }
}

// ============================================================
// RoPE: in-place on Q (with H^-0.5 scale); Kraw -> Kout (f32 output)
// and Kbf (bf16 for attention). grid = B*T, 128 threads.
// ============================================================
__global__ __launch_bounds__(128) void rope_kernel(
    float* __restrict__ Qb, const float* __restrict__ Kraw,
    float* __restrict__ Kout, unsigned short* __restrict__ Kbf,
    const int* __restrict__ positions)
{
    const int bt = blockIdx.x;
    const int h  = threadIdx.x;          // 0..127
    const float pos  = (float)positions[bt];
    const float freq = __expf(-(2.0f * (float)h / (float)HH) * __logf(10000.0f));
    float s, c;
    __sincosf(pos * freq, &s, &c);
    const float qs = 0.0625f;            // H^-0.5

    float* q = Qb + (long)bt * NH * HH;
    #pragma unroll
    for (int n = 0; n < NH; ++n) {
        float x1 = q[h], x2 = q[128 + h];
        q[h]       = (x1 * c - x2 * s) * qs;
        q[128 + h] = (x2 * c + x1 * s) * qs;
        q += HH;
    }
    const long kb = (long)bt * HH;
    float x1 = Kraw[kb + h], x2 = Kraw[kb + 128 + h];
    float k1 = x1 * c - x2 * s;
    float k2 = x2 * c + x1 * s;
    Kout[kb + h]       = k1;
    Kout[kb + 128 + h] = k2;
    Kbf[kb + h]        = f2bf(k1);
    Kbf[kb + 128 + h]  = f2bf(k2);
}

// ============================================================
// V transpose+convert: vout[b,t,h] f32 -> Vt[b,h,t] bf16
// ============================================================
__global__ __launch_bounds__(256) void vtrans_kernel(
    const float* __restrict__ V, unsigned short* __restrict__ Vt)
{
    const long i = (long)blockIdx.x * 256 + threadIdx.x;  // over B*T*H
    const int  h  = (int)(i & (HH - 1));
    const long bt = i >> 8;
    const int  t  = (int)(bt & (TT - 1));
    const int  b  = (int)(bt >> 10);
    Vt[((long)b * HH + h) * TT + t] = f2bf(V[i]);
}

// ============================================================
// Flash attention (causal, MQA): grid = (T/128, B*N), 256 thr.
// Q rows in register A-frags; bf16 K / V^T chunks staged into LDS
// with gfx1250 async global->LDS B128 copies; online softmax f32.
// ============================================================
__global__ __launch_bounds__(256) void attn_kernel(
    const float* __restrict__ Q, const unsigned short* __restrict__ Kbf,
    const unsigned short* __restrict__ Vbf, float* __restrict__ Enc)
{
    const int bn = blockIdx.y;
    const int b  = bn >> 3;
    const int n  = bn & 7;
    const int rowTile = blockIdx.x;

    __shared__ unsigned short Ks[32][264];     // [key][h]  row = 528B (16B mult)
    __shared__ unsigned short Vt[256][40];     // [h][key]  row = 80B  (16B mult)
    __shared__ unsigned short Pst[8][16][36];  // per-wave P restage

    const int tid  = threadIdx.x;
    const int lane = tid & 31;
    const int w    = tid >> 5;
    const int l16  = lane & 15;
    const int hi   = lane >> 4;

    const int r0 = rowTile * 128 + w * 16;

    // ---- load Q rows as 8 A-frags (f32 -> packed bf16) ----
    Frag aq[8];
    {
        const float* qrow = Q + (((long)(b * TT + r0 + l16)) * NH + n) * HH;
        const int lo = hi * 8;
        #pragma unroll
        for (int s = 0; s < 8; ++s) {
            const int h0 = s * 32;
            #pragma unroll
            for (int j = 0; j < 4; ++j) {
                int k0 = h0 + lo + 2 * j;
                aq[s].u[j]     = pack2bf(qrow[k0], qrow[k0 + 1]);
                int k1 = h0 + 16 + lo + 2 * j;
                aq[s].u[4 + j] = pack2bf(qrow[k1], qrow[k1 + 1]);
            }
        }
    }

    float mrow[8], lrow[8];
    v8f of[16];
    #pragma unroll
    for (int j = 0; j < 8; ++j) { mrow[j] = BIGNEG; lrow[j] = 0.0f; }
    #pragma unroll
    for (int c = 0; c < 16; ++c) of[c] = (v8f){};

    // async-copy source pointers for this thread
    const int kkey = tid >> 3;            // 0..31 (K stage: key row)
    const int kseg = (tid & 7) * 32;      // 32 halves = 64B segment
    const unsigned short* Kbase = Kbf + ((long)(b * TT + kkey) << 8) + kseg;
    const unsigned short* Vbase = Vbf + ((long)b * HH + tid) * TT;   // h = tid
    const unsigned ksDst = lds_addr32(&Ks[kkey][kseg]);
    const unsigned vtDst = lds_addr32(&Vt[tid][0]);

    const int kend = rowTile * 128 + 128;
    for (int kc = 0; kc < kend; kc += 32) {
        __syncthreads();                  // prev iter's LDS reads done
        {   // async global->LDS staging (bf16, no conversion in hot loop)
            const unsigned short* kg = Kbase + ((long)kc << 8);
            const unsigned short* vg = Vbase + kc;
            #pragma unroll
            for (int m = 0; m < 4; ++m) {
                async_ld_b128(ksDst + m * 16, kg + m * 8);
                async_ld_b128(vtDst + m * 16, vg + m * 8);
            }
        }
        wait_asynccnt0();
        __syncthreads();

        // ---- S = Q * K^T : two 16x16 frags ----
        v8f s0 = (v8f){}, s1 = (v8f){};
        {
            const int klo = hi * 16;
            #pragma unroll
            for (int s = 0; s < 8; ++s) {
                const int h0 = s * 32;
                Frag bk0, bk1;
                const unsigned short* kp0 = &Ks[l16][h0];
                const unsigned short* kp1 = &Ks[16 + l16][h0];
                #pragma unroll
                for (int j = 0; j < 8; ++j) {
                    bk0.u[j] = *(const unsigned int*)(kp0 + hi * 8 + 2 * j + (j >= 4 ? 8 : 0));
                    bk1.u[j] = *(const unsigned int*)(kp1 + hi * 8 + 2 * j + (j >= 4 ? 8 : 0));
                }
                s0 = wmma_bf16(aq[s].v, bk0.v, s0);
                s1 = wmma_bf16(aq[s].v, bk1.v, s1);
            }
        }

        // ---- causal mask + online softmax ----
        v8f svv;
        #pragma unroll
        for (int j = 0; j < 8; ++j) {
            const int row = r0 + hi * 8 + j;
            float e0 = ((kc + l16)      <= row) ? s0[j] : BIGNEG;
            float e1 = ((kc + 16 + l16) <= row) ? s1[j] : BIGNEG;
            float mx = fmaxf(e0, e1);
            #pragma unroll
            for (int d = 8; d >= 1; d >>= 1)
                mx = fmaxf(mx, __shfl_xor(mx, d, 32));
            const float mnew  = fmaxf(mrow[j], mx);
            const float scale = __expf(mrow[j] - mnew);
            const float p0 = __expf(e0 - mnew);
            const float p1 = __expf(e1 - mnew);
            s0[j] = p0; s1[j] = p1;
            float rs = p0 + p1;
            #pragma unroll
            for (int d = 8; d >= 1; d >>= 1)
                rs += __shfl_xor(rs, d, 32);
            lrow[j] = lrow[j] * scale + rs;
            mrow[j] = mnew;
            svv[j]  = scale;
        }
        #pragma unroll
        for (int c = 0; c < 16; ++c) of[c] = of[c] * svv;

        // ---- restage P (C-layout -> A-layout) through LDS ----
        #pragma unroll
        for (int j = 0; j < 8; ++j) {
            const int row = hi * 8 + j;
            Pst[w][row][l16]      = f2bf(s0[j]);
            Pst[w][row][16 + l16] = f2bf(s1[j]);
        }
        __syncthreads();

        // ---- O += P * V ----
        {
            Frag pa;
            const unsigned short* pp = &Pst[w][l16][0];
            const int lo  = hi * 8;
            const int klo = hi * 16;
            #pragma unroll
            for (int j = 0; j < 4; ++j) {
                pa.u[j]     = *(const unsigned int*)(pp + lo + 2 * j);
                pa.u[4 + j] = *(const unsigned int*)(pp + 16 + lo + 2 * j);
            }
            #pragma unroll
            for (int c = 0; c < 16; ++c) {
                Frag bv;
                const unsigned short* vp = &Vt[c * 16 + l16][0];
                #pragma unroll
                for (int j = 0; j < 8; ++j)
                    bv.u[j] = *(const unsigned int*)(vp + lo + 2 * j + (j >= 4 ? 8 : 0));
                of[c] = wmma_bf16(pa.v, bv.v, of[c]);
            }
        }
    }

    // ---- normalize + store enc[b,t,n,h] (incremental pointers) ----
    v8f inv;
    #pragma unroll
    for (int j = 0; j < 8; ++j) inv[j] = 1.0f / lrow[j];
    const int tb = r0 + hi * 8;
    #pragma unroll
    for (int c = 0; c < 16; ++c) {
        v8f o = of[c] * inv;
        float* ep = Enc + (((long)(b * TT + tb)) * NH + n) * HH + c * 16 + l16;
        #pragma unroll
        for (int j = 0; j < 8; ++j) {
            *ep = o[j];
            ep += NH * HH;
        }
    }
}

__global__ void idx_kernel(int* __restrict__ out) {
    if (threadIdx.x < 4) out[threadIdx.x] = TT;
}

// ============================================================
// host-side launch
// ============================================================
extern "C" void kernel_launch(void* const* d_in, const int* in_sizes, int n_in,
                              void* d_out, int out_size, void* d_ws, size_t ws_size,
                              hipStream_t stream) {
    const float* x0     = (const float*)d_in[0];
    const float* x1     = (const float*)d_in[1];
    const int*   pos    = (const int*)  d_in[2];
    const float* q0_w   = (const float*)d_in[4];
    const float* kv0_w  = (const float*)d_in[5];
    const float* out0_w = (const float*)d_in[6];
    const float* q1_w   = (const float*)d_in[7];
    const float* kv1_w  = (const float*)d_in[8];
    const float* out1_w = (const float*)d_in[9];

    float* outf = (float*)d_out;
    // output layout: out0 | out1 | idx | k | v
    const long OUT1 = (long)BB * S0 * W0;                 // 6291456
    const long IDXO = OUT1 + (long)BB * S1 * W1;          // 7340032
    const long KOUT = IDXO + BB;                          // 7340036
    const long VOUT = KOUT + (long)BB * TT * HH;          // 8388612

    float* wsf  = (float*)d_ws;
    float* Qbuf = wsf;                                    // B*T*N*H f32
    float* Kraw = wsf + (long)BB * TT * NH * HH;          // B*T*H  f32
    float* Enc  = Kraw + (long)BB * TT * HH;              // B*T*N*H f32
    unsigned short* Kbf  = (unsigned short*)(Enc + (long)BB * TT * NH * HH); // B*T*H bf16
    unsigned short* VtBf = Kbf + (long)BB * TT * HH;                         // B*H*T bf16

    float* kout = outf + KOUT;
    float* vout = outf + VOUT;

    // ---- QKV projections (bf16 WMMA GEMMs) ----
    gemm_bf16_kernel<<<dim3(4, 12, BB * NH), 256, 0, stream>>>(
        x0, q0_w, Qbuf, S0, HH, W0, W0, HH, NH * HH,
        (long)S0 * W0, (long)W0 * HH, (long)TT * NH * HH, HH, NH);
    gemm_bf16_kernel<<<dim3(4, 4, BB * NH), 256, 0, stream>>>(
        x1, q1_w, Qbuf + (long)S0 * NH * HH, S1, HH, W1, W1, HH, NH * HH,
        (long)S1 * W1, (long)W1 * HH, (long)TT * NH * HH, HH, NH);
    gemm_bf16_kernel<<<dim3(4, 12, BB), 256, 0, stream>>>(
        x0, kv0_w, Kraw, S0, HH, W0, W0, HH, HH,
        (long)S0 * W0, 0L, (long)TT * HH, 0L, 1);
    gemm_bf16_kernel<<<dim3(4, 12, BB), 256, 0, stream>>>(
        x0, kv0_w + (long)W0 * HH, vout, S0, HH, W0, W0, HH, HH,
        (long)S0 * W0, 0L, (long)TT * HH, 0L, 1);
    gemm_bf16_kernel<<<dim3(4, 4, BB), 256, 0, stream>>>(
        x1, kv1_w, Kraw + (long)S0 * HH, S1, HH, W1, W1, HH, HH,
        (long)S1 * W1, 0L, (long)TT * HH, 0L, 1);
    gemm_bf16_kernel<<<dim3(4, 4, BB), 256, 0, stream>>>(
        x1, kv1_w + (long)W1 * HH, vout + (long)S0 * HH, S1, HH, W1, W1, HH, HH,
        (long)S1 * W1, 0L, (long)TT * HH, 0L, 1);

    // ---- RoPE (q in-place + scale; k -> d_out f32 and ws bf16) ----
    rope_kernel<<<BB * TT, 128, 0, stream>>>(Qbuf, Kraw, kout, Kbf, pos);

    // ---- V transpose+convert for attention staging ----
    vtrans_kernel<<<(BB * TT * HH) / 256, 256, 0, stream>>>(vout, VtBf);

    // ---- flash attention ----
    attn_kernel<<<dim3(TT / 128, BB * NH), 256, 0, stream>>>(Qbuf, Kbf, VtBf, Enc);

    // ---- output projections ----
    gemm_bf16_kernel<<<dim3(W0 / 64, 12, BB), 256, 0, stream>>>(
        Enc, out0_w, outf, S0, W0, NH * HH, NH * HH, W0, W0,
        (long)TT * NH * HH, 0L, (long)S0 * W0, 0L, 1);
    gemm_bf16_kernel<<<dim3(W1 / 64, 4, BB), 256, 0, stream>>>(
        Enc + (long)S0 * NH * HH, out1_w, outf + OUT1, S1, W1, NH * HH, NH * HH, W1, W1,
        (long)TT * NH * HH, 0L, (long)S1 * W1, 0L, 1);

    // ---- idx ----
    idx_kernel<<<1, 4, 0, stream>>>((int*)d_out + IDXO);
}
